// Quantization_17403207483789
// MI455X (gfx1250) — matrix operand order repaired and checked by reading, not compiled, not run on hardware
//
#include <hip/hip_runtime.h>

// ---- problem constants (match reference) ----
#define N_TOK 32768
#define DIM   256
#define KCB   8192
#define NTILE (KCB / 16)
#define XSTR  264   // padded bf16 row stride in LDS: 264*2B=528B, 132 dwords -> bank step 4

typedef __attribute__((ext_vector_type(2)))  float  v2f;
typedef __attribute__((ext_vector_type(8)))  float  v8f;
typedef __attribute__((ext_vector_type(8)))  __bf16 v8bf;
typedef __attribute__((ext_vector_type(16))) __bf16 v16bf;

// -----------------------------------------------------------------------------
// K1: codebook[K,D] = E[K,D] @ W[D,D]^T   (full f32 via V_WMMA_F32_16X16X4_F32)
// One wave computes one 16x16 output tile; 64 k-steps of 4.
// -----------------------------------------------------------------------------
__global__ __launch_bounds__(256) void k_codebook(const float* __restrict__ E,
                                                  const float* __restrict__ W,
                                                  float* __restrict__ CB) {
  int wave = (blockIdx.x * blockDim.x + threadIdx.x) >> 5;  // 0..8191
  int lane = threadIdx.x & 31;
  int tk = wave >> 4;          // 0..511 : codebook row tile
  int ti = wave & 15;          // 0..15  : output column tile
  int k0 = tk * 16, i0 = ti * 16;
  int mr   = lane & 15;        // A: M row / B: N column (ISA f32 A/B layout)
  int kofs = (lane >> 4) << 1; // K offset of this lane-half: 0 or 2
  const float* erow = E + (size_t)(k0 + mr) * DIM;  // A row (reduction along row)
  const float* wrow = W + (size_t)(i0 + mr) * DIM;  // B = W^T col = W row
  v8f acc = {};
  for (int j = 0; j < DIM; j += 4) {
    v2f a, b;
    a.x = erow[j + kofs]; a.y = erow[j + kofs + 1];
    b.x = wrow[j + kofs]; b.y = wrow[j + kofs + 1];
    acc = __builtin_amdgcn_wmma_f32_16x16x4_f32(false, a, false, b,
                                                (short)0, acc, false, false);
  }
  // C/D layout: VGPR r -> M = r + 8*(lane>=16), N = lane&15
  int row = k0 + ((lane >> 4) << 3);
  int col = i0 + (lane & 15);
#pragma unroll
  for (int r = 0; r < 8; ++r)
    CB[(size_t)(row + r) * DIM + col] = acc[r];
}

// -----------------------------------------------------------------------------
// K1b: c2[k] = sum_d codebook[k,d]^2   (one wave per row)
// -----------------------------------------------------------------------------
__global__ __launch_bounds__(256) void k_c2(const float* __restrict__ CB,
                                            float* __restrict__ C2) {
  int w = (blockIdx.x * blockDim.x + threadIdx.x) >> 5;  // row 0..8191
  int lane = threadIdx.x & 31;
  const float* row = CB + (size_t)w * DIM;
  float s = 0.f;
  for (int j = lane; j < DIM; j += 32) { float v = row[j]; s += v * v; }
#pragma unroll
  for (int m = 16; m; m >>= 1) s += __shfl_xor(s, m, 32);
  if (lane == 0) C2[w] = s;
}

// -----------------------------------------------------------------------------
// K2: fused distance GEMM + argmin.  dist = x2 - 2*(x.c) + c2, never stored.
// Block: 256 threads (8 wave32), 128 tokens.  bf16 WMMA 16x16x32, f32 accum.
// A fragments (x tile) are register-resident for the whole K loop; the 16-code
// codebook tile is double-buffered in LDS (one barrier per tile, staging of
// tile kt+1 overlaps the 8 WMMAs of tile kt).
// -----------------------------------------------------------------------------
__global__ __launch_bounds__(256) void k_argmin(const float* __restrict__ X,
                                                const float* __restrict__ CBp,
                                                const float* __restrict__ C2,
                                                int* __restrict__ IDS) {
  __shared__ __bf16 xs[128 * XSTR];       // 67.6 KB  x tile (bf16)
  __shared__ float  x2s[128];
  __shared__ __bf16 cbs[2][16 * XSTR];    // 2 x 8.4 KB codebook tile (bf16)

  int t = threadIdx.x;
  int lane = t & 31, wv = t >> 5, h = lane >> 4;
  int tokBase = blockIdx.x * 128;

  // ---- stage x (f32 -> bf16 LDS) + per-row sum of squares; 2 threads/row ----
  {
    int r = t >> 1, half = t & 1;
    const float* xrow = X + (size_t)(tokBase + r) * DIM + half * 128;
    __bf16* dst = &xs[r * XSTR + half * 128];
    float ss = 0.f;
#pragma unroll 8
    for (int j = 0; j < 128; j += 4) {
      float4 v = *(const float4*)(xrow + j);
      ss += v.x * v.x + v.y * v.y + v.z * v.z + v.w * v.w;
      dst[j] = (__bf16)v.x; dst[j + 1] = (__bf16)v.y;
      dst[j + 2] = (__bf16)v.z; dst[j + 3] = (__bf16)v.w;
    }
    ss += __shfl_xor(ss, 1, 32);
    if (!half) x2s[r] = ss;
  }

  // cooperative codebook-tile stage: 16 rows f32 -> bf16, 16 floats/thread
  auto stage = [&](int tile, int buf) {
    int cr = t >> 4, cc = (t & 15) * 16;
    const float* src = CBp + (size_t)(tile * 16 + cr) * DIM + cc;
    __bf16* dst = &cbs[buf][cr * XSTR + cc];
#pragma unroll
    for (int j = 0; j < 16; j += 4) {
      float4 v = *(const float4*)(src + j);
      dst[j] = (__bf16)v.x; dst[j + 1] = (__bf16)v.y;
      dst[j + 2] = (__bf16)v.z; dst[j + 3] = (__bf16)v.w;
    }
  };

  stage(0, 0);                    // preload first tile
  __syncthreads();                // also covers the x staging above

  // ---- hoist A fragments into registers (invariant over the K loop) ----
  // A layout (16-bit, 16x32): lane-half h holds K = {h*8+0..7, 16+h*8+0..7}
  const __bf16* arow = &xs[(wv * 16 + (lane & 15)) * XSTR];  // A: M = lane&15
  v16bf afrag[8];
#pragma unroll
  for (int jj = 0; jj < 8; ++jj) {
    const __bf16* p = arow + jj * 32;
    v8bf alo = *(const v8bf*)(p + h * 8);
    v8bf ahi = *(const v8bf*)(p + 16 + h * 8);
    afrag[jj] = __builtin_shufflevector(alo, ahi, 0, 1, 2, 3, 4, 5, 6, 7,
                                        8, 9, 10, 11, 12, 13, 14, 15);
  }

  // this lane's 8 token rows (C/D layout: M = r + 8*h)
  float myx2[8];
#pragma unroll
  for (int r = 0; r < 8; ++r) myx2[r] = x2s[wv * 16 + h * 8 + r];

  float bestD[8]; int bestI[8];
#pragma unroll
  for (int r = 0; r < 8; ++r) { bestD[r] = 3.402823466e38f; bestI[r] = 0; }

  for (int kt = 0; kt < NTILE; ++kt) {
    int cur = kt & 1;
    if (kt + 1 < NTILE) stage(kt + 1, cur ^ 1);   // overlap with WMMAs below
    if (kt + 2 < NTILE && t < 16)                 // warm tile kt+2 in L2/L0
      __builtin_prefetch(CBp + (size_t)((kt + 2) * 16 + t) * DIM, 0, 0);

    // 8 x V_WMMA_F32_16X16X32_BF16 over D=256
    // B layout (32x16): lane-half h holds K = {h*8+0..7, 16+h*8+0..7} (matches A)
    const __bf16* brow = &cbs[cur][(lane & 15) * XSTR];
    v8f acc = {};
#pragma unroll
    for (int jj = 0; jj < 8; ++jj) {
      v8bf blo = *(const v8bf*)(brow + jj * 32 + h * 8);
      v8bf bhi = *(const v8bf*)(brow + jj * 32 + 16 + h * 8);
      v16bf b = __builtin_shufflevector(blo, bhi, 0, 1, 2, 3, 4, 5, 6, 7, 8, 9, 10, 11, 12, 13, 14, 15);
      acc = __builtin_amdgcn_wmma_f32_16x16x32_bf16(false, afrag[jj], false, b,
                                                    (short)0, acc, false, false);
    }

    int   n   = kt * 16 + (lane & 15);
    float c2n = C2[n];
#pragma unroll
    for (int r = 0; r < 8; ++r) {
      float d = myx2[r] - 2.0f * acc[r] + c2n;     // same formula as reference
      if (d < bestD[r]) { bestD[r] = d; bestI[r] = n; }  // kt ascending => first-min kept
    }
    __syncthreads();   // tile kt+1 fully staged; reads of buf cur complete
  }

  // argmin across the 16 lanes of each half (same tokens, different codes)
#pragma unroll
  for (int r = 0; r < 8; ++r) {
    float d = bestD[r]; int i = bestI[r];
#pragma unroll
    for (int m = 1; m <= 8; m <<= 1) {
      float d2 = __shfl_xor(d, m, 32);
      int   i2 = __shfl_xor(i, m, 32);
      if (d2 < d || (d2 == d && i2 < i)) { d = d2; i = i2; }  // lowest-index tie-break
    }
    if ((lane & 15) == 0) IDS[tokBase + wv * 16 + h * 8 + r] = i;
  }
}

// -----------------------------------------------------------------------------
// K3: quantized = codebook[ids] (f32), ids -> float, per-block loss partials
// -----------------------------------------------------------------------------
__global__ __launch_bounds__(256) void k_gather(const float* __restrict__ X,
                                                const float* __restrict__ CB,
                                                const int* __restrict__ IDS,
                                                float* __restrict__ outQ,
                                                float* __restrict__ outI,
                                                float* __restrict__ PART) {
  __shared__ float red[256];
  int t = threadIdx.x;
  int tok = blockIdx.x * 128 + (t >> 1);
  int half = t & 1;
  int id = IDS[tok];
  const float* q  = CB + (size_t)id * DIM + half * 128;
  const float* xr = X + (size_t)tok * DIM + half * 128;
  float* o = outQ + (size_t)tok * DIM + half * 128;
  float ss = 0.f;
#pragma unroll 8
  for (int j = 0; j < 128; j += 4) {
    float4 qv = *(const float4*)(q + j);
    float4 xv = *(const float4*)(xr + j);
    *(float4*)(o + j) = qv;
    float a = xv.x - qv.x, b = xv.y - qv.y, c = xv.z - qv.z, d = xv.w - qv.w;
    ss += a * a + b * b + c * c + d * d;
  }
  if (!half) outI[tok] = (float)id;
  red[t] = ss;
  __syncthreads();
  for (int s = 128; s > 0; s >>= 1) {   // fixed-order tree -> deterministic
    if (t < s) red[t] += red[t + s];
    __syncthreads();
  }
  if (t == 0) PART[blockIdx.x] = red[0];
}

// -----------------------------------------------------------------------------
// K4: final loss = 1.25 * sum / (N*D)   (256 partials, fixed-order reduce)
// -----------------------------------------------------------------------------
__global__ __launch_bounds__(256) void k_loss(const float* __restrict__ PART,
                                              float* __restrict__ outL) {
  __shared__ float red[256];
  int t = threadIdx.x;
  red[t] = PART[t];
  __syncthreads();
  for (int s = 128; s > 0; s >>= 1) {
    if (t < s) red[t] += red[t + s];
    __syncthreads();
  }
  if (t == 0) *outL = red[0] * (1.25f / 8388608.0f);  // 1.25 / (N*D)
}

// -----------------------------------------------------------------------------
extern "C" void kernel_launch(void* const* d_in, const int* in_sizes, int n_in,
                              void* d_out, int out_size, void* d_ws, size_t ws_size,
                              hipStream_t stream) {
  (void)in_sizes; (void)n_in; (void)out_size; (void)ws_size;
  const float* x    = (const float*)d_in[0];   // [N, D]
  const float* emb  = (const float*)d_in[1];   // [K, D]
  const float* wout = (const float*)d_in[2];   // [D, D]

  // workspace layout (~8.55 MB)
  float* CB   = (float*)d_ws;                  // [K, D] f32 codebook
  float* C2   = CB + (size_t)KCB * DIM;        // [K]
  int*   IDS  = (int*)(C2 + KCB);              // [N]
  float* PART = (float*)(IDS + N_TOK);         // [256]

  float* outQ = (float*)d_out;                 // [N*D]
  float* outI = outQ + (size_t)N_TOK * DIM;    // [N] ids as float
  float* outL = outI + N_TOK;                  // [1] loss

  k_codebook<<<dim3(1024), dim3(256), 0, stream>>>(emb, wout, CB);
  k_c2      <<<dim3(1024), dim3(256), 0, stream>>>(CB, C2);
  k_argmin  <<<dim3(N_TOK / 128), dim3(256), 0, stream>>>(x, CB, C2, IDS);
  k_gather  <<<dim3(N_TOK / 128), dim3(256), 0, stream>>>(x, CB, IDS, outQ, outI, PART);
  k_loss    <<<dim3(1), dim3(256), 0, stream>>>(PART, outL);
}
